// BasicBlock_41025527611475
// MI455X (gfx1250) — matrix operand order repaired and checked by reading, not compiled, not run on hardware
//
#include <hip/hip_runtime.h>
#include <hip/hip_bf16.h>

typedef __attribute__((ext_vector_type(16))) _Float16 v16h;
typedef __attribute__((ext_vector_type(8)))  _Float16 v8h;
typedef __attribute__((ext_vector_type(8)))  float    v8f;
typedef __attribute__((ext_vector_type(4)))  unsigned v4u;
typedef __attribute__((ext_vector_type(8)))  unsigned v8u;

#define CH    128
#define HDIM  56
#define WDIM  56
#define HWPIX 3136                 // 56*56
#define NPIX  100352               // 32*3136
#define NPC   12845056L            // NPIX*128
#define WELEM 147456               // 128*128*9

// per-lane async copy: global -> LDS, tracked by ASYNCcnt (ISA 08 §4)
#define ASYNC_CP16(ldsoff, gaddr) \
    asm volatile("global_load_async_to_lds_b128 %0, %1, off" \
                 :: "v"(ldsoff), "v"(gaddr) : "memory")
#define ASYNC_CP16_OFS16(ldsoff, gaddr) \
    asm volatile("global_load_async_to_lds_b128 %0, %1, off offset:16" \
                 :: "v"(ldsoff), "v"(gaddr) : "memory")
#define WAIT_ASYNC_LE2()  asm volatile("s_wait_asynccnt 0x2" ::: "memory")
#define WAIT_ASYNC_0()    asm volatile("s_wait_asynccnt 0x0" ::: "memory")
#define WAIT_TENSOR_LE1() asm volatile("s_wait_tensorcnt 0x1" ::: "memory")
#define WAIT_TENSOR_0()   asm volatile("s_wait_tensorcnt 0x0" ::: "memory")

__device__ __forceinline__ unsigned lds_off(const void* p) {
    // generic LDS address: low 32 bits are the LDS byte offset (ISA §10.2)
    return (unsigned)(unsigned long long)p;
}

// ---------------------------------------------------------------------------
// 0) zero BN stat accumulators (2 x {sum[128], sumsq[128]}) + 32-half zero pad
// ---------------------------------------------------------------------------
__global__ void k_zero_stats(float* stats) {
    int t = threadIdx.x;
    if (t < 528) stats[t] = 0.0f;   // 512 stats + 16 floats (32 halfs) zero pad
}

// ---------------------------------------------------------------------------
// 1) LSQ-quantize both weights and repack to [tap(9)][kc(4)][N=128][K=32] f16
//    forward alpha_s == alpha;  wq = rint(clip(w/alpha,-4,3)) * alpha
// ---------------------------------------------------------------------------
__global__ __launch_bounds__(256) void k_pack_weights(
    const float* __restrict__ w1, const float* __restrict__ alpha1,
    const float* __restrict__ w2, const float* __restrict__ alpha2,
    _Float16* __restrict__ w1p, _Float16* __restrict__ w2p)
{
    int i = blockIdx.x * 256 + threadIdx.x;
    if (i >= 2 * WELEM) return;
    int sel = (i >= WELEM);
    int l = i - sel * WELEM;
    const float* w = sel ? w2 : w1;
    float alpha    = sel ? alpha2[0] : alpha1[0];
    _Float16* dst  = sel ? w2p : w1p;

    int cout = l / 1152;            // src: [cout][cin][kh][kw]
    int r    = l - cout * 1152;
    int cin  = r / 9;
    int kk   = r - cin * 9;
    int kh = kk / 3, kw = kk - kh * 3;

    float q = rintf(fminf(fmaxf(w[l] / alpha, -4.0f), 3.0f)) * alpha;

    int tap = kh * 3 + kw;
    int kc  = cin >> 5;
    dst[(((tap * 4 + kc) * 128 + cout) << 5) + (cin & 31)] = (_Float16)q;
}

// ---------------------------------------------------------------------------
// 2) x: NCHW fp32 -> NHWC f16
// ---------------------------------------------------------------------------
__global__ __launch_bounds__(256) void k_nchw_to_nhwc_f16(
    const float* __restrict__ x, _Float16* __restrict__ xh)
{
    long o = ((long)blockIdx.x * 256 + threadIdx.x) * 8;
    if (o >= NPC) return;
    int p  = (int)(o >> 7);
    int c0 = (int)(o & 127);
    int b  = p / HWPIX;
    int hw = p - b * HWPIX;
    _Float16 buf[8];
    #pragma unroll
    for (int j = 0; j < 8; ++j)
        buf[j] = (_Float16)x[(long)(b * CH + c0 + j) * HWPIX + hw];
    *(uint4*)(xh + o) = *(const uint4*)buf;
}

// ---------------------------------------------------------------------------
// 3) implicit-GEMM 3x3 conv via v_wmma_f32_16x16x32_f16.
//    A tiles: per-lane global_load_async_to_lds_b128 (ASYNCcnt pipeline).
//    W tiles: Tensor Data Mover tensor_load_to_lds, one D# per stage issued
//             by wave 0 (TENSORcnt pipeline).  Double-buffered LDS.
//    Epilogue transposes D fragments through LDS for coalesced b128 stores.
// ---------------------------------------------------------------------------
__global__ __launch_bounds__(256) void k_conv3x3_wmma(
    const _Float16* __restrict__ xin,   // NHWC f16 [NPIX][128]
    const _Float16* __restrict__ wp,    // [9][4][128][32] f16
    const _Float16* __restrict__ zpad,  // 32 halfs of zeros
    _Float16* __restrict__ yout,        // NHWC f16 [NPIX][128]
    float* __restrict__ stats)          // [256] = sum[128], sumsq[128]
{
    // overlay: stages use smem as A[2][4096] | W[2][4096]; epilogue reuses the
    // whole 32KB as a 128x128 f16 output tile.
    __shared__ _Float16 smem[16384];
    __shared__ float ldsSum[128];
    __shared__ float ldsSS[128];

    const int t    = threadIdx.x;
    const int lane = t & 31;
    const int wv   = t >> 5;
    const int m    = lane & 15;
    const int hi   = lane >> 4;
    const int pg   = blockIdx.x * 128;

    if (t < 128) { ldsSum[t] = 0.0f; ldsSS[t] = 0.0f; }

    // staging role: thread t -> pixel t/2, 16-half channel slice (t&1)*16
    const int lp = t >> 1;
    const int ch = (t & 1) * 16;
    const int p  = pg + lp;
    const int b  = p / HWPIX;
    const int hw = p - b * HWPIX;
    const int h  = hw / WDIM;
    const int w  = hw - h * WDIM;

    // activation tile: 2 x 16B per thread, per-lane async (handles zero-pad)
    auto issueA = [&](int s, int buf) {
        const int tap = s >> 2, kc = s & 3;
        const int kh = tap / 3 - 1;
        const int kw = tap - (tap / 3) * 3 - 1;
        const int hs = h + kh, vs = w + kw;
        const bool valid = (hs >= 0) & (hs < HDIM) & (vs >= 0) & (vs < WDIM);
        const _Float16* gA = valid
            ? (xin + ((long)b * HWPIX + hs * WDIM + vs) * 128 + kc * 32 + ch)
            : zpad;
        unsigned           la = lds_off(&smem[buf * 4096 + lp * 32 + ch]);
        unsigned long long ga = (unsigned long long)gA;
        ASYNC_CP16(la, ga);
        ASYNC_CP16_OFS16(la, ga);
    };

    // weight tile: one TDM descriptor moves the whole 8KB tile (wave 0 only)
    auto issueW = [&](int s, int buf) {
        unsigned long long gw = (unsigned long long)(wp + ((long)s << 12));
        unsigned lw = lds_off(&smem[8192 + buf * 4096]);
        v4u g0 = { 1u,                                    // count=1, user D#
                   lw,                                    // lds_addr
                   (unsigned)gw,                          // global_addr[31:0]
                   (unsigned)(gw >> 32) | 0x80000000u };  // ga[56:32] | type=2
        v8u g1 = { 1u << 16,     // data_size = 2 bytes
                   4096u << 16,  // tensor_dim0 = 4096 (lo 16 in [31:16])
                   1u << 16,     // tensor_dim0 hi=0 | tensor_dim1 = 1
                   4096u << 16,  // tensor_dim1 hi=0 | tile_dim0 = 4096
                   1u,           // tile_dim1 = 1, tile_dim2 = 0
                   4096u,        // tensor_dim0_stride = 4096
                   0u, 0u };     // stride0 hi, tensor_dim1_stride = 0
        asm volatile("tensor_load_to_lds %0, %1" :: "s"(g0), "s"(g1) : "memory");
        if (s + 2 < 36)   // warm GL2 for the stage after next
            __builtin_prefetch(wp + ((s + 2) << 12), 0, 1);
    };

    v8f acc[8];
    #pragma unroll
    for (int i = 0; i < 8; ++i) {
        v8f z = {0.f, 0.f, 0.f, 0.f, 0.f, 0.f, 0.f, 0.f};
        acc[i] = z;
    }

    issueA(0, 0);
    issueA(1, 1);
    if (wv == 0) { issueW(0, 0); issueW(1, 1); }

    for (int s = 0; s < 36; ++s) {
        // stage s resident (in-order completion), stage s+1 still in flight
        if (s < 35) { WAIT_ASYNC_LE2(); WAIT_TENSOR_LE1(); }
        else        { WAIT_ASYNC_0();   WAIT_TENSOR_0();   }
        __syncthreads();

        const int buf = s & 1;
        // A fragment (16x32 f16): row M=m; K = 16*hi + j
        const _Float16* ap = &smem[buf * 4096 + (wv * 16 + m) * 32 + hi * 16];
        v8h alo = *(const v8h*)(ap);
        v8h ahi = *(const v8h*)(ap + 8);
        v16h a = __builtin_shufflevector(alo, ahi,
                 0, 1, 2, 3, 4, 5, 6, 7, 8, 9, 10, 11, 12, 13, 14, 15);

        const _Float16* wb = &smem[8192 + buf * 4096];
        #pragma unroll
        for (int nt = 0; nt < 8; ++nt) {
            // B fragment (32x16 f16): column N=m, K = 16*hi + j
            v16h bf = *(const v16h*)(wb + (nt * 16 + m) * 32 + hi * 16);
            acc[nt] = __builtin_amdgcn_wmma_f32_16x16x32_f16(
                false, a, false, bf, (short)0, acc[nt], false, false);
        }
        __syncthreads();                 // everyone done reading buf
        if (s + 2 < 36) {
            issueA(s + 2, buf);
            if (wv == 0) issueW(s + 2, buf);
        }
    }

    // epilogue: BN stats from registers + transpose through LDS
    // D layout: pixel-in-tile = wv*16 + 8*hi + r, channel = nt*16 + m
    #pragma unroll
    for (int nt = 0; nt < 8; ++nt) {
        const int c = nt * 16 + m;
        float s1 = 0.f, ss = 0.f;
        #pragma unroll
        for (int r = 0; r < 8; ++r) {
            float v = acc[nt][r];
            s1 += v; ss += v * v;
            smem[(wv * 16 + hi * 8 + r) * 128 + c] = (_Float16)v;
        }
        atomicAdd(&ldsSum[c], s1);
        atomicAdd(&ldsSS[c], ss);
    }
    __syncthreads();

    // coalesced writeout: thread t -> pixel t/2, 64-half slice (t&1)*64
    {
        const int lp2 = t >> 1;
        const int c0  = (t & 1) * 64;
        const _Float16* src = smem + lp2 * 128 + c0;
        _Float16* dst = yout + (long)(pg + lp2) * 128 + c0;
        #pragma unroll
        for (int j = 0; j < 8; ++j)
            *(uint4*)(dst + j * 8) = *(const uint4*)(src + j * 8);
    }
    if (t < 128) {
        atomicAdd(&stats[t], ldsSum[t]);
        atomicAdd(&stats[128 + t], ldsSS[t]);
    }
}

// ---------------------------------------------------------------------------
// 4) BN (batch stats) + ReLU, f16 NHWC -> f16 NHWC
// ---------------------------------------------------------------------------
__global__ __launch_bounds__(256) void k_bn_relu(
    const _Float16* __restrict__ y, const float* __restrict__ stats,
    const float* __restrict__ gamma, const float* __restrict__ beta,
    _Float16* __restrict__ out)
{
    long o = ((long)blockIdx.x * 256 + threadIdx.x) * 8;
    if (o >= NPC) return;
    int c0 = (int)(o & 127);
    const float inv_n = 1.0f / (float)NPIX;
    _Float16 buf[8];
    *(uint4*)buf = *(const uint4*)(y + o);
    _Float16 ob[8];
    #pragma unroll
    for (int j = 0; j < 8; ++j) {
        int c = c0 + j;
        float mean = stats[c] * inv_n;
        float var  = stats[128 + c] * inv_n - mean * mean;
        float sc   = gamma[c] * rsqrtf(var + 1e-5f);
        float bi   = beta[c] - mean * sc;
        float v    = (float)buf[j] * sc + bi;
        ob[j] = (_Float16)fmaxf(v, 0.0f);
    }
    *(uint4*)(out + o) = *(const uint4*)ob;
}

// ---------------------------------------------------------------------------
// 5) BN2 + residual + ReLU -> fp32 NCHW output
// ---------------------------------------------------------------------------
__global__ __launch_bounds__(256) void k_bn_add_relu_out(
    const _Float16* __restrict__ y2, const float* __restrict__ stats,
    const float* __restrict__ gamma, const float* __restrict__ beta,
    const float* __restrict__ x, float* __restrict__ out)
{
    long idx = ((long)blockIdx.x * 256 + threadIdx.x) * 4;  // NCHW flat
    if (idx >= NPC) return;
    int b   = (int)(idx / (CH * HWPIX));
    int rem = (int)(idx - (long)b * CH * HWPIX);
    int c   = rem / HWPIX;
    int hw  = rem - c * HWPIX;
    long p  = (long)b * HWPIX + hw;

    const float inv_n = 1.0f / (float)NPIX;
    float mean = stats[c] * inv_n;
    float var  = stats[128 + c] * inv_n - mean * mean;
    float sc   = gamma[c] * rsqrtf(var + 1e-5f);
    float bi   = beta[c] - mean * sc;

    float4 xres = *(const float4*)(x + idx);
    float o0 = (float)y2[(p + 0) * 128 + c] * sc + bi + xres.x;
    float o1 = (float)y2[(p + 1) * 128 + c] * sc + bi + xres.y;
    float o2 = (float)y2[(p + 2) * 128 + c] * sc + bi + xres.z;
    float o3 = (float)y2[(p + 3) * 128 + c] * sc + bi + xres.w;
    float4 r = {fmaxf(o0, 0.f), fmaxf(o1, 0.f), fmaxf(o2, 0.f), fmaxf(o3, 0.f)};
    *(float4*)(out + idx) = r;
}

// ---------------------------------------------------------------------------
extern "C" void kernel_launch(void* const* d_in, const int* in_sizes, int n_in,
                              void* d_out, int out_size, void* d_ws, size_t ws_size,
                              hipStream_t stream) {
    const float* x      = (const float*)d_in[0];
    const float* w1     = (const float*)d_in[1];
    const float* alpha1 = (const float*)d_in[2];
    const float* gamma1 = (const float*)d_in[3];
    const float* beta1  = (const float*)d_in[4];
    const float* w2     = (const float*)d_in[5];
    const float* alpha2 = (const float*)d_in[6];
    const float* gamma2 = (const float*)d_in[7];
    const float* beta2  = (const float*)d_in[8];
    float* out = (float*)d_out;

    // workspace layout (y2h aliases xh: xh dead once conv1 consumers finish)
    _Float16* xh   = (_Float16*)d_ws;     // NPC halfs (also y2h)
    _Float16* y1h  = xh  + NPC;           // NPC halfs
    _Float16* a1   = y1h + NPC;           // NPC halfs
    _Float16* w1p  = a1  + NPC;           // WELEM halfs
    _Float16* w2p  = w1p + WELEM;         // WELEM halfs
    float*    stats = (float*)(w2p + WELEM);     // 512 floats + zero pad
    _Float16* zpad  = (_Float16*)(stats + 512);  // 32 halfs of zeros
    _Float16* y2h  = xh;

    k_zero_stats<<<1, 544, 0, stream>>>(stats);
    k_pack_weights<<<(2 * WELEM + 255) / 256, 256, 0, stream>>>(
        w1, alpha1, w2, alpha2, w1p, w2p);
    k_nchw_to_nhwc_f16<<<(int)(NPC / 8 / 256), 256, 0, stream>>>(x, xh);

    k_conv3x3_wmma<<<NPIX / 128, 256, 0, stream>>>(xh, w1p, zpad, y1h, stats);
    k_bn_relu<<<(int)(NPC / 8 / 256), 256, 0, stream>>>(
        y1h, stats, gamma1, beta1, a1);
    k_conv3x3_wmma<<<NPIX / 128, 256, 0, stream>>>(a1, w2p, zpad, y2h, stats + 256);
    k_bn_add_relu_out<<<(int)(NPC / 4 / 256), 256, 0, stream>>>(
        y2h, stats + 256, gamma2, beta2, x, out);
}
